// Model_5523327942836
// MI455X (gfx1250) — matrix operand order repaired and checked by reading, not compiled, not run on hardware
//
#include <hip/hip_runtime.h>
#include <hip/hip_bf16.h>

typedef __attribute__((ext_vector_type(16))) _Float16 v16h;
typedef __attribute__((ext_vector_type(8)))  float    v8f;

namespace {
constexpr int kB = 512, kN = 32, kTin = 12, kHid = 64, kHeads = 8, kOut = 9, kDeg = 8;
constexpr int kF1    = kHeads * kHid;   // 512
constexpr int kEperB = kN * kDeg;       // 256
constexpr int kEtot  = kB * kEperB;     // 131072
constexpr int kNE    = kEperB + kN;     // 288 incl. self loops
constexpr int kG3    = 3 * kHid;        // 192

// ---- dynamic LDS layout (bytes); all fragment buffers 32B aligned ----
constexpr int OFF_H1   = 0;        // 32x512 f32 (pre-agg GAT1 activations)      64KB
constexpr int OFF_H1BF = 65536;    // h1 as f16 B-fragments (K=32,N=512)         32KB
constexpr int OFF_H1AF = 98304;    // elu(h1) as f16 A-fragments (M=32,K=512)    32KB
constexpr int OFF_SCR  = 131072;   // 64KB scratch: W1Bf | Am(f32) | AmAf | W2Bf
constexpr int OFF_XSA  = 196608;   // x as f16 A-fragments (M=32,K=32)            2KB
constexpr int OFF_ES   = 198656;   // 32x8 f32
constexpr int OFF_ED   = 199680;
constexpr int OFF_MM   = 200704;
constexpr int OFF_SS   = 201728;
constexpr int OFF_SRC  = 202752;   // 288 int
constexpr int OFF_DST  = 203904;
constexpr int OFF_H2   = 205056;   // 32x64 f32
constexpr int OFF_H2BF = 213248;   // h2 as f16 B-fragments (K=32,N=64)           4KB
constexpr int OFF_ES2  = 217344;
constexpr int OFF_ED2  = 217472;
constexpr int OFF_MM2  = 217600;
constexpr int OFF_SS2  = 217728;
constexpr int OFF_A2   = 217856;   // 32x32 f32
constexpr int OFF_A2AF = 221952;   // A2 as f16 A-fragments                       2KB
constexpr int OFF_SEQ  = 224000;   // 64x32 f32 GRU input
constexpr int OFF_HS1  = 232192;
constexpr int OFF_HS2  = 232448;
constexpr int OFF_GH   = 232704;
constexpr int OFF_XT2  = 233472;
constexpr int OFF_GH2  = 234240;
constexpr int SMEM_BYTES = 235008; // ~229.5KB < 320KB/WGP
// phase-overlapped regions
constexpr int OFF_W1BF = OFF_SCR;           // 32KB, dead after GEMM1
constexpr int OFF_AM   = OFF_SCR + 32768;   // 8x32x32 f32, dead after AmAf built
constexpr int OFF_AMAF = OFF_SCR;           // 16KB, dead after aggregation1
constexpr int OFF_W2BF = OFF_SCR;           // 64KB, GAT2 phase
constexpr int OFF_XI   = OFF_H1BF;          // 48KB GRU xi overlays dead h1 frags
}

union CF { v8f v; float f[8]; };

__device__ __forceinline__ v8f wmma16(v16h a, v16h b, v8f c) {
  return __builtin_amdgcn_wmma_f32_16x16x32_f16(false, a, false, b, (short)0, c, false, false);
}
__device__ __forceinline__ v16h ldfrag(const _Float16* p) {       // 32B LDS load
  return *(const v16h*)p;                                         // -> 2x ds_load_b128
}
__device__ __forceinline__ float eluf(float v)  { return v > 0.f ? v : (__expf(v) - 1.f); }
__device__ __forceinline__ float lrelu(float v) { return v > 0.f ? v : 0.2f * v; }
__device__ __forceinline__ float sigm(float v)  { return 1.f / (1.f + __expf(-v)); }
// A-fragment K index (contiguous: k = 16*g + j, matches B fragments)
__device__ __forceinline__ int akidx(int j, int g) { return j + 16 * g; }

__global__ __launch_bounds__(256, 1)
void stgat_gru_fused(const float* __restrict__ x,   const int* __restrict__ ei,
                     const float* __restrict__ W1,  const float* __restrict__ as1,
                     const float* __restrict__ ad1, const float* __restrict__ b1,
                     const float* __restrict__ W2,  const float* __restrict__ as2,
                     const float* __restrict__ ad2, const float* __restrict__ b2,
                     const float* __restrict__ gam, const float* __restrict__ bet,
                     const float* __restrict__ Wi1, const float* __restrict__ Wh1,
                     const float* __restrict__ bi1, const float* __restrict__ bh1,
                     const float* __restrict__ Wi2, const float* __restrict__ Wh2,
                     const float* __restrict__ bi2, const float* __restrict__ bh2,
                     const float* __restrict__ Wf,  const float* __restrict__ bf,
                     float* __restrict__ out)
{
  extern __shared__ char smem[];
  float*    h1   = (float*)   (smem + OFF_H1);
  _Float16* h1bf = (_Float16*)(smem + OFF_H1BF);
  _Float16* h1af = (_Float16*)(smem + OFF_H1AF);
  _Float16* w1bf = (_Float16*)(smem + OFF_W1BF);
  _Float16* w2bf = (_Float16*)(smem + OFF_W2BF);
  _Float16* amaf = (_Float16*)(smem + OFF_AMAF);
  _Float16* xsa  = (_Float16*)(smem + OFF_XSA);
  float*    Am   = (float*)   (smem + OFF_AM);
  float*    es   = (float*)   (smem + OFF_ES);
  float*    ed   = (float*)   (smem + OFF_ED);
  float*    mm   = (float*)   (smem + OFF_MM);
  float*    ss   = (float*)   (smem + OFF_SS);
  int*      esrc = (int*)     (smem + OFF_SRC);
  int*      edst = (int*)     (smem + OFF_DST);
  float*    h2   = (float*)   (smem + OFF_H2);
  _Float16* h2bf = (_Float16*)(smem + OFF_H2BF);
  float*    es2  = (float*)   (smem + OFF_ES2);
  float*    ed2  = (float*)   (smem + OFF_ED2);
  float*    mm2  = (float*)   (smem + OFF_MM2);
  float*    ss2  = (float*)   (smem + OFF_SS2);
  float*    A2   = (float*)   (smem + OFF_A2);
  _Float16* a2af = (_Float16*)(smem + OFF_A2AF);
  float*    seq  = (float*)   (smem + OFF_SEQ);
  float*    xi   = (float*)   (smem + OFF_XI);
  float*    hs1  = (float*)   (smem + OFF_HS1);
  float*    hs2  = (float*)   (smem + OFF_HS2);
  float*    gh   = (float*)   (smem + OFF_GH);
  float*    xt2  = (float*)   (smem + OFF_XT2);
  float*    gh2  = (float*)   (smem + OFF_GH2);

  const int tid  = threadIdx.x;
  const int b    = blockIdx.x;
  const int lane = tid & 31;
  const int w    = tid >> 5;          // wave 0..7
  const int g    = lane >> 4;         // half-wave 0/1
  const int ln   = lane & 15;

  if (tid < 32) __builtin_prefetch(W2 + tid * 1024, 0, 1);  // warm L2 (global_prefetch_b8)

  // ---- stage x as A-fragments (zero-padded K=12->32) ----
  for (int idx = tid; idx < kN * 32; idx += 256) {          // idx = ((m*2+g)*16+j)
    const int m = idx >> 5, gg = (idx >> 4) & 1, j = idx & 15;
    const int k = akidx(j, gg);
    xsa[idx] = (k < kTin) ? (_Float16)x[b * kN * kTin + m * kTin + k] : (_Float16)0.f;
  }
  // ---- stage W1 as B-fragments (zero-padded rows) ----
  for (int idx = tid; idx < kF1 * 32; idx += 256) {         // idx = ((n*2+g)*16+t)
    const int n = idx >> 5, gg = (idx >> 4) & 1, t = idx & 15;
    const int k = t + 16 * gg;
    w1bf[idx] = (k < kTin) ? (_Float16)W1[k * kF1 + n] : (_Float16)0.f;
  }
  __syncthreads();

  // ========== GAT1 GEMM: h1[32,512] = x @ W1 ; 64 tiles, 8 per wave ==========
  for (int it = 0; it < 8; ++it) {
    const int tile = w + 8 * it;
    const int mt = tile >> 5, nt = tile & 31;
    const v16h a  = ldfrag(xsa  + ((mt * 16 + ln) * 2 + g) * 16);
    const v16h bb = ldfrag(w1bf + ((nt * 16 + ln) * 2 + g) * 16);
    CF c; c.v = v8f{};
    c.v = wmma16(a, bb, c.v);
    const int col = nt * 16 + ln;
    #pragma unroll
    for (int r = 0; r < 8; ++r) {
      const int row = mt * 16 + 8 * g + r;
      const float v = c.f[r];
      h1[row * kF1 + col] = v;                               // f32 for logits
      h1bf[(col * 2 + mt) * 16 + 8 * g + r] = (_Float16)v;   // f16 B-frag for aggregation
    }
  }
  __syncthreads();

  // ---- attention logits es/ed per (node, head) ----
  {
    const int n = tid >> 3, hh = tid & 7;
    float se = 0.f, sd = 0.f;
    #pragma unroll 4
    for (int c = 0; c < kHid; ++c) {
      const float hv = h1[n * kF1 + hh * kHid + c];
      se += hv * as1[hh * kHid + c];
      sd += hv * ad1[hh * kHid + c];
    }
    es[n * kHeads + hh] = se;
    ed[n * kHeads + hh] = sd;
  }
  // ---- load batch-local edges (+ self loops) ----
  {
    const int e = b * kEperB + tid;
    esrc[tid] = ei[e] - b * kN;
    edst[tid] = ei[kEtot + e] - b * kN;
    if (tid < kN) { esrc[kEperB + tid] = tid; edst[kEperB + tid] = tid; }
  }
  __syncthreads();

  // ---- per-dst softmax stats (fixed scan order => deterministic) ----
  {
    const int d = tid >> 3, hh = tid & 7;
    const float edv = ed[d * kHeads + hh];
    float m = -1e30f;
    for (int e = 0; e < kNE; ++e)
      if (edst[e] == d) m = fmaxf(m, lrelu(es[esrc[e] * kHeads + hh] + edv));
    float s = 0.f;
    for (int e = 0; e < kNE; ++e)
      if (edst[e] == d) s += __expf(lrelu(es[esrc[e] * kHeads + hh] + edv) - m);
    mm[d * kHeads + hh] = m;
    ss[d * kHeads + hh] = s + 1e-16f;
  }
  for (int i = tid; i < kHeads * kN * kN; i += 256) Am[i] = 0.f;
  __syncthreads();
  // ---- build per-head alpha matrices (owner-thread rows), then f16 A-frags ----
  {
    const int d = tid >> 3, hh = tid & 7;
    const float edv = ed[d * kHeads + hh];
    const float m = mm[d * kHeads + hh], inv = 1.f / ss[d * kHeads + hh];
    for (int e = 0; e < kNE; ++e)
      if (edst[e] == d) {
        const int s = esrc[e];
        Am[hh * kN * kN + d * kN + s] +=
            __expf(lrelu(es[s * kHeads + hh] + edv) - m) * inv;
      }
  }
  __syncthreads();   // Am complete; W1Bf (same region as AmAf) no longer needed
  {
    const int d = tid >> 3, hh = tid & 7;
    #pragma unroll
    for (int gg = 0; gg < 2; ++gg)
      #pragma unroll
      for (int j = 0; j < 16; ++j)
        amaf[((hh * kN + d) * 2 + gg) * 16 + j] =
            (_Float16)Am[hh * kN * kN + d * kN + akidx(j, gg)];
  }
  __syncthreads();

  // ==== GAT1 aggregation WMMA: out_h = A_h[32x32] @ h_head[32x64] (64 tiles) ====
  // reads AmAf/h1Bf, writes elu(out+b1) straight into GAT2 A-fragments (no hazard)
  for (int it = 0; it < 8; ++it) {
    const int tile = w + 8 * it;
    const int hh = tile >> 3, mt = (tile >> 2) & 1, nt = tile & 3;
    const v16h a  = ldfrag(amaf + ((hh * kN + mt * 16 + ln) * 2 + g) * 16);
    const int col = hh * kHid + nt * 16 + ln;
    const v16h bb = ldfrag(h1bf + (col * 2 + g) * 16);
    CF c; c.v = v8f{};
    c.v = wmma16(a, bb, c.v);
    const int kb = col >> 5, kc = col & 31;
    const int gA = kc >> 4, jA = kc & 15;
    #pragma unroll
    for (int r = 0; r < 8; ++r) {
      const int row = mt * 16 + 8 * g + r;
      h1af[((kb * kN + row) * 2 + gA) * 16 + jA] = (_Float16)eluf(c.f[r] + b1[col]);
    }
  }
  __syncthreads();

  // ---- stage W2 as B-fragments (overwrites scratch; AmAf dead) ----
  for (int idx = tid; idx < kF1 * kHid; idx += 256) {  // idx = ((kb*64+n)*2+g)*16+t
    const int kb = idx >> 11, rem = idx & 2047;
    const int n = rem >> 5, gg = (rem >> 4) & 1, t = rem & 15;
    w2bf[idx] = (_Float16)W2[(kb * 32 + t + 16 * gg) * kHid + n];
  }
  __syncthreads();

  // ========== GAT2 GEMM: h2[32,64] = elu(h1) @ W2 ; 1 tile/wave, 16 K-steps ==========
  {
    const int mt = (w >> 2) & 1, nt = w & 3;
    CF c; c.v = v8f{};
    for (int kb = 0; kb < 16; ++kb) {
      const v16h a  = ldfrag(h1af + ((kb * kN + mt * 16 + ln) * 2 + g) * 16);
      const v16h bb = ldfrag(w2bf + ((kb * kHid + nt * 16 + ln) * 2 + g) * 16);
      c.v = wmma16(a, bb, c.v);
    }
    const int col = nt * 16 + ln;
    #pragma unroll
    for (int r = 0; r < 8; ++r) {
      const int row = mt * 16 + 8 * g + r;
      h2[row * kHid + col] = c.f[r];
      h2bf[(col * 2 + mt) * 16 + 8 * g + r] = (_Float16)c.f[r];
    }
  }
  __syncthreads();

  // ---- GAT2 attention (1 head) ----
  if (tid < kN) {
    float se = 0.f, sd = 0.f;
    #pragma unroll 4
    for (int c = 0; c < kHid; ++c) {
      const float hv = h2[tid * kHid + c];
      se += hv * as2[c]; sd += hv * ad2[c];
    }
    es2[tid] = se; ed2[tid] = sd;
  }
  __syncthreads();
  if (tid < kN) {
    const int d = tid; const float edv = ed2[d];
    float m = -1e30f;
    for (int e = 0; e < kNE; ++e)
      if (edst[e] == d) m = fmaxf(m, lrelu(es2[esrc[e]] + edv));
    float s = 0.f;
    for (int e = 0; e < kNE; ++e)
      if (edst[e] == d) s += __expf(lrelu(es2[esrc[e]] + edv) - m);
    mm2[d] = m; ss2[d] = s + 1e-16f;
    for (int sN = 0; sN < kN; ++sN) A2[d * kN + sN] = 0.f;
    const float inv = 1.f / ss2[d];
    for (int e = 0; e < kNE; ++e)
      if (edst[e] == d)
        A2[d * kN + esrc[e]] += __expf(lrelu(es2[esrc[e]] + edv) - m) * inv;
    #pragma unroll
    for (int gg = 0; gg < 2; ++gg)
      #pragma unroll
      for (int j = 0; j < 16; ++j)
        a2af[((d) * 2 + gg) * 16 + j] = (_Float16)A2[d * kN + akidx(j, gg)];
  }
  __syncthreads();

  // ==== GAT2 aggregation WMMA (8 tiles) + bias + ELU; writes f32 h2 (reads h2Bf) ====
  {
    const int mt = (w >> 2) & 1, nt = w & 3;
    const v16h a  = ldfrag(a2af + ((mt * 16 + ln) * 2 + g) * 16);
    const v16h bb = ldfrag(h2bf + ((nt * 16 + ln) * 2 + g) * 16);
    CF c; c.v = v8f{};
    c.v = wmma16(a, bb, c.v);
    const int col = nt * 16 + ln;
    #pragma unroll
    for (int r = 0; r < 8; ++r)
      h2[(mt * 16 + 8 * g + r) * kHid + col] = eluf(c.f[r] + b2[col]);
  }
  __syncthreads();

  // ---- LayerNorm over 64 channels; transpose into seq[time][feat] ----
  if (tid < kN) {
    float mu = 0.f;
    for (int c = 0; c < kHid; ++c) mu += h2[tid * kHid + c];
    mu *= (1.f / kHid);
    float var = 0.f;
    for (int c = 0; c < kHid; ++c) { const float dv = h2[tid * kHid + c] - mu; var += dv * dv; }
    var *= (1.f / kHid);
    const float rstd = rsqrtf(var + 1e-5f);
    for (int c = 0; c < kHid; ++c)
      seq[c * kN + tid] = (h2[tid * kHid + c] - mu) * rstd * gam[c] + bet[c];
  }
  __syncthreads();

  // ---- GRU1 input gates precompute (xi overlays dead h1 fragment buffers) ----
  for (int idx = tid; idx < kHid * kG3; idx += 256) {
    const int t = idx / kG3, j = idx % kG3;
    float s = bi1[j];
    #pragma unroll 4
    for (int f = 0; f < kN; ++f) s += seq[t * kN + f] * Wi1[j * kN + f];
    xi[t * kG3 + j] = s;
  }
  if (tid < kHid) { hs1[tid] = 0.f; hs2[tid] = 0.f; }
  __syncthreads();

  // ---- fused GRU1 + GRU2 recurrence (64 serial steps) ----
  for (int t = 0; t < kHid; ++t) {
    if (tid < kG3) {
      float s = bh1[tid];
      #pragma unroll 4
      for (int c = 0; c < kHid; ++c) s += hs1[c] * Wh1[tid * kHid + c];
      gh[tid] = s;
    }
    __syncthreads();
    if (tid < kHid) {
      const float r = sigm(xi[t * kG3 + tid] + gh[tid]);
      const float z = sigm(xi[t * kG3 + kHid + tid] + gh[kHid + tid]);
      const float n = tanhf(xi[t * kG3 + 2 * kHid + tid] + r * gh[2 * kHid + tid]);
      hs1[tid] = (1.f - z) * n + z * hs1[tid];
    }
    __syncthreads();
    if (tid < kG3) {
      float si = bi2[tid], sh = bh2[tid];
      #pragma unroll 4
      for (int c = 0; c < kHid; ++c) {
        si += hs1[c] * Wi2[tid * kHid + c];
        sh += hs2[c] * Wh2[tid * kHid + c];
      }
      xt2[tid] = si; gh2[tid] = sh;
    }
    __syncthreads();
    if (tid < kHid) {
      const float r = sigm(xt2[tid] + gh2[tid]);
      const float z = sigm(xt2[kHid + tid] + gh2[kHid + tid]);
      const float n = tanhf(xt2[2 * kHid + tid] + r * gh2[2 * kHid + tid]);
      hs2[tid] = (1.f - z) * n + z * hs2[tid];
    }
    __syncthreads();
  }

  // ---- final FC ----
  if (tid < kOut) {
    float s = bf[tid];
    #pragma unroll 4
    for (int c = 0; c < kHid; ++c) s += hs2[c] * Wf[tid * kHid + c];
    out[b * kOut + tid] = s;
  }
}

extern "C" void kernel_launch(void* const* d_in, const int* in_sizes, int n_in,
                              void* d_out, int out_size, void* d_ws, size_t ws_size,
                              hipStream_t stream) {
  (void)in_sizes; (void)n_in; (void)out_size; (void)d_ws; (void)ws_size;
  const float* x   = (const float*)d_in[0];
  const int*   ei  = (const int*)  d_in[1];
  const float* W1  = (const float*)d_in[2];
  const float* as1 = (const float*)d_in[3];
  const float* ad1 = (const float*)d_in[4];
  const float* b1  = (const float*)d_in[5];
  const float* W2  = (const float*)d_in[6];
  const float* as2 = (const float*)d_in[7];
  const float* ad2 = (const float*)d_in[8];
  const float* b2  = (const float*)d_in[9];
  const float* gam = (const float*)d_in[10];
  const float* bet = (const float*)d_in[11];
  const float* Wi1 = (const float*)d_in[12];
  const float* Wh1 = (const float*)d_in[13];
  const float* bi1 = (const float*)d_in[14];
  const float* bh1 = (const float*)d_in[15];
  const float* Wi2 = (const float*)d_in[16];
  const float* Wh2 = (const float*)d_in[17];
  const float* bi2 = (const float*)d_in[18];
  const float* bh2 = (const float*)d_in[19];
  const float* Wf  = (const float*)d_in[20];
  const float* bf  = (const float*)d_in[21];
  float* out = (float*)d_out;

  stgat_gru_fused<<<dim3(kB), dim3(256), SMEM_BYTES, stream>>>(
      x, ei, W1, as1, ad1, b1, W2, as2, ad2, b2, gam, bet,
      Wi1, Wh1, bi1, bh1, Wi2, Wh2, bi2, bh2, Wf, bf, out);
}
